// AttentionWithCoverage_19052474925157
// MI455X (gfx1250) — compile-verified, matrix operand order checked by reading, not run on hardware
//
#include <hip/hip_runtime.h>
#include <hip/hip_bf16.h>

#define B_   32
#define L_   4096
#define AD   512   // ATTN_DIM
#define ED   512   // ENC_DIM
#define DD   256   // DEC_DIM
#define KS   11
#define KPAD 544   // ED + 32 (Fcov slot padded to one 32-wide k-step)
#define FPAD 32

typedef __attribute__((ext_vector_type(16))) __bf16 v16bf;
typedef __attribute__((ext_vector_type(8)))  float  v8f;

// ---------------- async copy helpers (CDNA5 async-to-LDS path) ----------------

#if defined(__has_builtin)
#  if __has_builtin(__builtin_amdgcn_global_load_async_to_lds_b128)
#    define HAVE_ASYNC_BUILTIN 1
#  endif
#  if __has_builtin(__builtin_amdgcn_s_wait_asynccnt)
#    define HAVE_ASYNCCNT_BUILTIN 1
#  endif
#endif

// exact param type per hipcc diagnostic: '__attribute__((vector_size(16))) int AS1 *'
typedef __attribute__((vector_size(16))) int i32x4;
typedef __attribute__((address_space(1))) i32x4* gv4_ptr;
typedef __attribute__((address_space(3))) i32x4* lv4_ptr;

__device__ __forceinline__ void async_cp16(const void* gsrc, void* ldst) {
#if defined(HAVE_ASYNC_BUILTIN)
  gv4_ptr g = (gv4_ptr)(unsigned long long)gsrc;
  lv4_ptr l = (lv4_ptr)(unsigned)(unsigned long long)ldst;   // low 32 bits = LDS addr
  __builtin_amdgcn_global_load_async_to_lds_b128(g, l, 0, 0);
#else
  unsigned           laddr = (unsigned)(unsigned long long)ldst;
  unsigned long long gaddr = (unsigned long long)gsrc;
  asm volatile("global_load_async_to_lds_b128 %0, %1, off"
               :: "v"(laddr), "v"(gaddr) : "memory");
#endif
}

__device__ __forceinline__ void async_wait0() {
#if defined(HAVE_ASYNCCNT_BUILTIN)
  __builtin_amdgcn_s_wait_asynccnt(0);
#else
  asm volatile("s_wait_asynccnt 0" ::: "memory");
#endif
}

// copy one 17408-byte Wcat slab (1088 x 16B) to LDS, partitioned over 256 threads
__device__ __forceinline__ void prefetch_slab(const __bf16* src, __bf16* dst, int tid) {
  const char* s = (const char*)src;
  char*       d = (char*)dst;
#pragma unroll
  for (int j = 0; j < 4; ++j) {
    int off = (tid + j * 256) * 16;
    async_cp16(s + off, d + off);
  }
  if (tid < 64) {
    int off = (tid + 1024) * 16;
    async_cp16(s + off, d + off);
  }
}

// ---------------- small prep kernels ----------------

// h_t = h_prev @ W_a^T : (B, AD)
__global__ void ht_kernel(const float* __restrict__ h_prev,
                          const float* __restrict__ W_a,
                          float* __restrict__ ht) {
  int idx = blockIdx.x * 256 + threadIdx.x;        // B*AD = 16384
  int b = idx >> 9, a = idx & (AD - 1);
  float s = 0.f;
  for (int k = 0; k < DD; ++k) s += h_prev[b * DD + k] * W_a[a * DD + k];
  ht[idx] = s;
}

// Wcat bf16 (AD, KPAD): [U_a | U_f | 0-pad]
__global__ void wcat_kernel(const float* __restrict__ U_a,
                            const float* __restrict__ U_f,
                            __bf16* __restrict__ wcat) {
  int idx = blockIdx.x * 256 + threadIdx.x;        // AD*KPAD = 278528
  if (idx >= AD * KPAD) return;
  int a = idx / KPAD, k = idx % KPAD;
  float v = (k < ED) ? U_a[a * ED + k]
          : (k < ED + KS) ? U_f[a * KS + (k - ED)] : 0.f;
  wcat[idx] = (__bf16)v;
}

// Coverage conv -> Fcov bf16 (B, L, FPAD); cross-correlation, 'same' pad
__global__ void fcov_kernel(const float* __restrict__ beta_prev,
                            const float* __restrict__ conv_w,
                            __bf16* __restrict__ fcov) {
  int idx = blockIdx.x * 256 + threadIdx.x;        // B*L = 131072
  int b = idx >> 12, l = idx & (L_ - 1);
  float x[KS];
#pragma unroll
  for (int t = 0; t < KS; ++t) {
    int p = l + t - (KS - 1) / 2;
    x[t] = (p >= 0 && p < L_) ? beta_prev[b * L_ + p] : 0.f;
  }
  __bf16* o = fcov + (size_t)idx * FPAD;
#pragma unroll
  for (int oc = 0; oc < KS; ++oc) {
    float s = 0.f;
#pragma unroll
    for (int t = 0; t < KS; ++t) s += conv_w[oc * KS + t] * x[t];
    o[oc] = (__bf16)s;
  }
#pragma unroll
  for (int oc = KS; oc < FPAD; ++oc) o[oc] = (__bf16)0.f;
}

// ---------------- main fused energy kernel (BF16 WMMA + async double-buffer) ----------------
// grid (B, L/128), block 256 (8 waves); wave w owns rows l0 = blk*128 + w*16
__global__ void __launch_bounds__(256)
energy_kernel(const float* __restrict__ ann,
              const float* __restrict__ ht,
              const __bf16* __restrict__ wcat,
              const __bf16* __restrict__ fcov,
              const float* __restrict__ va,
              float* __restrict__ energy) {
  const int tid  = threadIdx.x;
  const int b    = blockIdx.x;
  const int wave = tid >> 5;
  const int lane = tid & 31;
  const int half = lane >> 4;          // 0: K groups {0..7,16..23}; 1: {8..15,24..31}
  const int lm   = lane & 15;          // A-frag row (M) / B-frag col (N)
  const int l0   = blockIdx.y * 128 + wave * 16;

  __shared__ __bf16 lds_w[2][16 * KPAD];   // double-buffered weight slab (2 x 17408 B)

  // ---- load all 17 A-fragments once (annotations read exactly once) ----
  const float*  arow = ann  + ((size_t)b * L_ + (l0 + lm)) * ED;
  const __bf16* frow = fcov + ((size_t)b * L_ + (l0 + lm)) * FPAD;
  v16bf afrag[17];
#pragma unroll
  for (int ks = 0; ks < 16; ++ks) {
    const int kb = ks * 32 + half * 8;
    v16bf a;
#pragma unroll
    for (int i = 0; i < 8; ++i) a[i]     = (__bf16)arow[kb + i];
#pragma unroll
    for (int i = 0; i < 8; ++i) a[8 + i] = (__bf16)arow[kb + 16 + i];
    afrag[ks] = a;
  }
  {
    const int kb = half * 8;
    v16bf a;
#pragma unroll
    for (int i = 0; i < 8; ++i) a[i]     = frow[kb + i];
#pragma unroll
    for (int i = 0; i < 8; ++i) a[8 + i] = frow[kb + 16 + i];
    afrag[16] = a;
  }

  // prime the pipeline: async-prefetch slab 0
  prefetch_slab(wcat, &lds_w[0][0], tid);

  float eacc[8];
#pragma unroll
  for (int r = 0; r < 8; ++r) eacc[r] = 0.f;

  for (int nt = 0; nt < AD / 16; ++nt) {           // 32 N-tiles over attn dim
    async_wait0();          // my slab-nt chunks landed in LDS
    __syncthreads();        // everyone's landed; everyone done reading slab nt-1
    if (nt + 1 < AD / 16)
      prefetch_slab(wcat + (size_t)(nt + 1) * 16 * KPAD,
                    &lds_w[(nt + 1) & 1][0], tid);

    const __bf16* wrow = &lds_w[nt & 1][0] + lm * KPAD;  // row a = nt*16 + lm
    v8f c0 = {}, c1 = {};                                // dual chains for ILP
#pragma unroll
    for (int ks = 0; ks < 17; ++ks) {
      const int kb = ks * 32 + half * 8;
      v16bf bf;
#pragma unroll
      for (int i = 0; i < 8; ++i) bf[i]     = wrow[kb + i];
#pragma unroll
      for (int i = 0; i < 8; ++i) bf[8 + i] = wrow[kb + 16 + i];
      if (ks & 1)
        c1 = __builtin_amdgcn_wmma_f32_16x16x32_bf16(
               false, afrag[ks], false, bf, (short)0, c1, false, false);
      else
        c0 = __builtin_amdgcn_wmma_f32_16x16x32_bf16(
               false, afrag[ks], false, bf, (short)0, c0, false, false);
    }

    const int aidx = nt * 16 + lm;
    const float h = ht[b * AD + aidx];
    const float v = va[aidx];
#pragma unroll
    for (int r = 0; r < 8; ++r) eacc[r] += tanhf(c0[r] + c1[r] + h) * v;
  }

  // reduce over N (16 lanes of each half-wave)
#pragma unroll
  for (int r = 0; r < 8; ++r) {
    float s = eacc[r];
#pragma unroll
    for (int m = 1; m < 16; m <<= 1) s += __shfl_xor(s, m, 32);
    eacc[r] = s;
  }
  if (lm == 0) {
#pragma unroll
    for (int r = 0; r < 8; ++r)
      energy[b * L_ + l0 + half * 8 + r] = eacc[r];  // C/D: M = 8*half + r
  }
}

// ---------------- softmax + beta ----------------
__global__ void softmax_kernel(const float* __restrict__ energy,
                               const float* __restrict__ beta_prev,
                               float* __restrict__ alpha_out,
                               float* __restrict__ beta_out) {
  int b = blockIdx.x;
  __shared__ float red[256];
  const float* e = energy + (size_t)b * L_;
  float m = -INFINITY;
  for (int i = threadIdx.x; i < L_; i += 256) m = fmaxf(m, e[i]);
  red[threadIdx.x] = m; __syncthreads();
  for (int s = 128; s > 0; s >>= 1) {
    if (threadIdx.x < s) red[threadIdx.x] = fmaxf(red[threadIdx.x], red[threadIdx.x + s]);
    __syncthreads();
  }
  m = red[0]; __syncthreads();
  float sum = 0.f;
  for (int i = threadIdx.x; i < L_; i += 256) sum += __expf(e[i] - m);
  red[threadIdx.x] = sum; __syncthreads();
  for (int s = 128; s > 0; s >>= 1) {
    if (threadIdx.x < s) red[threadIdx.x] += red[threadIdx.x + s];
    __syncthreads();
  }
  sum = red[0]; __syncthreads();
  float inv = 1.f / sum;
  for (int i = threadIdx.x; i < L_; i += 256) {
    float a = __expf(e[i] - m) * inv;
    alpha_out[(size_t)b * L_ + i] = a;
    beta_out[(size_t)b * L_ + i] = beta_prev[(size_t)b * L_ + i] + a;
  }
}

// ---------------- context (two-stage, deterministic) ----------------
__global__ void ctx_part_kernel(const float* __restrict__ ann,
                                const float* __restrict__ alpha,
                                float* __restrict__ part) {
  int b = blockIdx.x, seg = blockIdx.y;            // 16 segments of 256 rows
  int d = threadIdx.x;                             // 512 threads
  float acc = 0.f;
  int l0 = seg * (L_ / 16);
  for (int l = l0; l < l0 + L_ / 16; ++l)
    acc += alpha[(size_t)b * L_ + l] * ann[((size_t)b * L_ + l) * ED + d];
  part[((size_t)b * 16 + seg) * ED + d] = acc;
}

__global__ void ctx_reduce_kernel(const float* __restrict__ part,
                                  float* __restrict__ ctx) {
  int idx = blockIdx.x * 256 + threadIdx.x;        // B*ED = 16384
  int b = idx >> 9, d = idx & (ED - 1);
  float s = 0.f;
#pragma unroll
  for (int seg = 0; seg < 16; ++seg) s += part[((size_t)b * 16 + seg) * ED + d];
  ctx[idx] = s;
}

// ---------------- launcher ----------------
extern "C" void kernel_launch(void* const* d_in, const int* in_sizes, int n_in,
                              void* d_out, int out_size, void* d_ws, size_t ws_size,
                              hipStream_t stream) {
  const float* annotations = (const float*)d_in[0];
  const float* h_prev      = (const float*)d_in[1];
  const float* beta_prev   = (const float*)d_in[2];
  const float* W_a         = (const float*)d_in[3];
  const float* U_a         = (const float*)d_in[4];
  const float* U_f         = (const float*)d_in[5];
  const float* v_a         = (const float*)d_in[6];
  const float* conv_w      = (const float*)d_in[7];

  float* out   = (float*)d_out;
  float* ctx   = out;                       // (B, ED)
  float* alpha = out + (size_t)B_ * ED;     // (B, L)
  float* beta  = alpha + (size_t)B_ * L_;   // (B, L)

  char* ws = (char*)d_ws;
  float*  ht    = (float*)(ws);                              // 64 KB
  __bf16* wcat  = (__bf16*)(ws + 65536);                     // 557056 B
  __bf16* fcov  = (__bf16*)(ws + 622592);                    // 8 MB
  float*  energ = (float*)(ws + 9011200);                    // 512 KB
  float*  cpart = (float*)(ws + 9535488);                    // 1 MB

  ht_kernel<<<B_ * AD / 256, 256, 0, stream>>>(h_prev, W_a, ht);
  wcat_kernel<<<(AD * KPAD + 255) / 256, 256, 0, stream>>>(U_a, U_f, wcat);
  fcov_kernel<<<B_ * L_ / 256, 256, 0, stream>>>(beta_prev, conv_w, fcov);

  dim3 eg(B_, L_ / 128);
  energy_kernel<<<eg, 256, 0, stream>>>(annotations, ht, wcat, fcov, v_a, energ);

  softmax_kernel<<<B_, 256, 0, stream>>>(energ, beta_prev, alpha, beta);

  dim3 cg(B_, 16);
  ctx_part_kernel<<<cg, 512, 0, stream>>>(annotations, alpha, cpart);
  ctx_reduce_kernel<<<B_ * ED / 256, 256, 0, stream>>>(cpart, ctx);
}
